// Attention_31052613550177
// MI455X (gfx1250) — compile-verified, matrix operand order checked by reading, not run on hardware
//
#include <hip/hip_runtime.h>
#include <hip/hip_bf16.h>

// ---------------------------------------------------------------------------
// MI455X (gfx1250) self-attention, B=8, N=H*W=4096 tokens, C=256.
// All GEMMs (QKV proj, QK^T, PV, out proj) run on v_wmma_f32_16x16x32_bf16.
// K/V streaming in the attention kernel uses the Tensor Data Mover
// (tensor_load_to_lds, TENSORcnt) with double buffering and DMA-inserted LDS
// row padding for conflict-free ds_load_b128 fragment reads.
// ---------------------------------------------------------------------------

typedef __attribute__((ext_vector_type(16))) __bf16 v16bf;
typedef __attribute__((ext_vector_type(8)))  __bf16 v8bf;
typedef __attribute__((ext_vector_type(8)))  float  v8f;
typedef __attribute__((ext_vector_type(4)))  float  v4f;
typedef __attribute__((ext_vector_type(4)))  unsigned int v4u;
typedef __attribute__((ext_vector_type(8)))  int    v8i;
typedef __attribute__((ext_vector_type(4)))  int    v4i;

union ABfrag { v16bf v; v8bf h[2]; };

static __device__ __forceinline__ v8f vzero8() {
    v8f z = {0.f,0.f,0.f,0.f,0.f,0.f,0.f,0.f};
    return z;
}

#define WMMA_BF16(A, B, C) \
    __builtin_amdgcn_wmma_f32_16x16x32_bf16(false, (A), false, (B), (short)0, (C), false, false)

#define C_DIM   256
#define N_TOK   4096      // tokens per batch (64*64)
#define B_DIM   8
#define TOK_ALL 32768     // B * N

// Workspace layout (bf16 elements):
//   WT  : 4 * 256*256 transposed bf16 weights (q,k,v,o)
//   Q   : 32768 x 256 row-major
//   K   : 32768 x 256 row-major
//   Vt  : per-batch transposed V: [b][c][tok]
//   O   : 32768 x 256 row-major attention output
#define WT_ELEMS  (4 * 65536)
#define MAT_ELEMS (TOK_ALL * C_DIM)   // 8388608

// ---------------------------------------------------------------------------
// TDM: issue a 2-D tensor_load_to_lds.  Dimensions/strides in 4-byte units
// (data_size=2 -> 4B elements).  dw0_flags carries data_size + pad config.
// D# group layouts per CDNA5 ISA §8.3/8.4; groups 2/3 zero (2-D tensor,
// no iteration, no gather, not in a cluster).
// ---------------------------------------------------------------------------
static __device__ __forceinline__ void tdm_load_2d(
    const void* gptr, unsigned lds_off, unsigned dw0_flags,
    unsigned tensor_d0, unsigned tensor_d1,
    unsigned tile_d0, unsigned tile_d1, unsigned stride0) {
    const unsigned long long ga = (unsigned long long)(size_t)gptr;
    v4u g0;
    g0[0] = 1u;                                    // count=1, user load descriptor
    g0[1] = lds_off;                               // LDS byte address of tile
    g0[2] = (unsigned)ga;                          // global_addr[31:0]
    g0[3] = (unsigned)(ga >> 32) | (2u << 30);     // global_addr[56:32] | type=2
    v8i g1;
    g1[0] = (int)dw0_flags;                        // mask=0 | data_size | pad cfg
    g1[1] = (int)(tensor_d0 << 16);                // tensor_dim0[15:0] at [63:48]
    g1[2] = (int)((tensor_d0 >> 16) | (tensor_d1 << 16));
    g1[3] = (int)((tensor_d1 >> 16) | (tile_d0 << 16));
    g1[4] = (int)(tile_d1 & 0xFFFFu);              // tile_dim1 | tile_dim2=0
    g1[5] = (int)stride0;                          // tensor_dim0_stride[31:0]
    g1[6] = 0;                                     // stride0[47:32] | stride1 lo
    g1[7] = 0;                                     // stride1 hi
    v4i z4 = {0, 0, 0, 0};
#if defined(__clang_major__) && __clang_major__ >= 23
    v8i z8 = {0, 0, 0, 0, 0, 0, 0, 0};
    __builtin_amdgcn_tensor_load_to_lds(g0, g1, z4, z4, z8, 0);
#else
    __builtin_amdgcn_tensor_load_to_lds(g0, g1, z4, z4, 0);
#endif
}

// dw0 flag fields: data_size=2 (4B) <<16, pad_enable<<20, pad_interval<<22,
// pad_amount<<25.  K rows: 128 dwords + 4 dword pad (interval code 6 = 128
// dwords, amount code 3 = 4 dwords).  V rows: 16 dwords + 4 dword pad
// (interval code 3 = 16 dwords).
#define TDM_FLAGS_K ((2u << 16) | (1u << 20) | (6u << 22) | (3u << 25))
#define TDM_FLAGS_V ((2u << 16) | (1u << 20) | (3u << 22) | (3u << 25))

// ---------------------------------------------------------------------------
// Kernel 0: transpose + convert weights to bf16.  wt[mat][d][c] = w[mat][c][d]
// ---------------------------------------------------------------------------
__global__ void k_prep_weights(const float* __restrict__ wq, const float* __restrict__ wk,
                               const float* __restrict__ wv, const float* __restrict__ wo,
                               __bf16* __restrict__ wt) {
    int idx = blockIdx.x * 256 + threadIdx.x;          // 262144 threads total
    int mat = idx >> 16;
    int rem = idx & 65535;
    int d = rem >> 8;
    int c = rem & 255;
    const float* src = (mat == 0) ? wq : (mat == 1) ? wk : (mat == 2) ? wv : wo;
    wt[idx] = (__bf16)src[c * 256 + d];
}

// ---------------------------------------------------------------------------
// Kernel 1: fused LayerNorm + Q/K/V projections.
// Block = 256 threads (8 waves), 64 tokens per block.
// ---------------------------------------------------------------------------
__global__ __launch_bounds__(256, 1)
void k_ln_qkv(const float* __restrict__ x,
              const float* __restrict__ gamma, const float* __restrict__ beta,
              const __bf16* __restrict__ wt,
              const float* __restrict__ bq, const float* __restrict__ bk,
              const float* __restrict__ bv,
              __bf16* __restrict__ Q, __bf16* __restrict__ K,
              __bf16* __restrict__ Vt) {
    __shared__ __align__(16) __bf16 hA[64][264];   // 528 B row stride (16B aligned)

    const int tid  = threadIdx.x;
    const int w    = tid >> 5;
    const int lane = tid & 31;
    const int blockbase = blockIdx.x * 64;

    // ---- LayerNorm: each wave handles 8 tokens, 8 channels per lane ----
    float g8[8], b8[8];
#pragma unroll
    for (int j = 0; j < 8; ++j) {
        g8[j] = gamma[lane * 8 + j];
        b8[j] = beta[lane * 8 + j];
    }
#pragma unroll
    for (int i = 0; i < 8; ++i) {
        const int tl = w * 8 + i;
        const float* xr = x + (size_t)(blockbase + tl) * C_DIM + lane * 8;
        v4f a0 = *(const v4f*)xr;
        v4f a1 = *(const v4f*)(xr + 4);
        float s = 0.f, q = 0.f;
#pragma unroll
        for (int j = 0; j < 4; ++j) { s += a0[j]; q += a0[j] * a0[j]; }
#pragma unroll
        for (int j = 0; j < 4; ++j) { s += a1[j]; q += a1[j] * a1[j]; }
#pragma unroll
        for (int m = 1; m < 32; m <<= 1) {
            s += __shfl_xor(s, m, 32);
            q += __shfl_xor(q, m, 32);
        }
        const float mean = s * (1.f / 256.f);
        const float var  = q * (1.f / 256.f) - mean * mean;
        const float rstd = rsqrtf(var + 1e-5f);
#pragma unroll
        for (int j = 0; j < 4; ++j)
            hA[tl][lane * 8 + j]     = (__bf16)((a0[j] - mean) * rstd * g8[j] + b8[j]);
#pragma unroll
        for (int j = 0; j < 4; ++j)
            hA[tl][lane * 8 + 4 + j] = (__bf16)((a1[j] - mean) * rstd * g8[4 + j] + b8[4 + j]);
    }
    __syncthreads();

    // ---- WMMA GEMM: h (64x256) x W (256x256) for q,k,v ----
    const int mt   = (w & 3) * 16;
    const int nh   = (w >> 2) * 128;
    const int lm   = lane & 15;
    const int g16  = lane >> 4;
    const int koff = g16 * 8;
    const int row  = mt + lm;

    const float* biases[3] = {bq, bk, bv};
#pragma unroll 1
    for (int proj = 0; proj < 3; ++proj) {
        const __bf16* wm = wt + proj * 65536;
        v8f acc[8];
#pragma unroll
        for (int nt = 0; nt < 8; ++nt) acc[nt] = vzero8();

#pragma unroll 1
        for (int ks = 0; ks < 8; ++ks) {
            const int kb = ks * 32 + koff;
            ABfrag a;
            a.h[0] = *(const v8bf*)&hA[row][kb];
            a.h[1] = *(const v8bf*)&hA[row][kb + 16];
#pragma unroll
            for (int nt = 0; nt < 8; ++nt) {
                const int d = nh + nt * 16 + lm;
                ABfrag b;
                b.h[0] = *(const v8bf*)(wm + d * 256 + kb);
                b.h[1] = *(const v8bf*)(wm + d * 256 + kb + 16);
                acc[nt] = WMMA_BF16(a.v, b.v, acc[nt]);
            }
        }
        const float* bias = biases[proj];
#pragma unroll
        for (int nt = 0; nt < 8; ++nt) {
            const int d   = nh + nt * 16 + lm;
            const float bsv = bias[d];
#pragma unroll
            for (int r = 0; r < 8; ++r) {
                const int m    = mt + r + 8 * g16;
                const int gtok = blockbase + m;
                const float val = acc[nt][r] + bsv;
                if (proj == 0) {
                    Q[(size_t)gtok * C_DIM + d] = (__bf16)val;
                } else if (proj == 1) {
                    K[(size_t)gtok * C_DIM + d] = (__bf16)val;
                } else {
                    const int bb = gtok >> 12;
                    const int tk = gtok & 4095;
                    Vt[((size_t)bb * C_DIM + d) * N_TOK + tk] = (__bf16)val;
                }
            }
        }
    }
}

// ---------------------------------------------------------------------------
// Kernel 2: flash attention with TDM-staged K/V.
// Block = 256 threads (8 waves); each wave owns 16 queries and streams all
// 4096 keys of its batch in chunks of 32.  Wave 0 drives the Tensor Data
// Mover: chunk i+1 DMAs into the alternate LDS buffer while all 8 waves run
// 32 WMMAs on chunk i.  TDM pad_enable inserts a 4-dword pad per row so the
// B-fragment ds_load_b128 reads are LDS-bank-conflict free.
// ---------------------------------------------------------------------------
__global__ __launch_bounds__(256, 1)
void k_attn(const __bf16* __restrict__ Q, const __bf16* __restrict__ K,
            const __bf16* __restrict__ Vt, __bf16* __restrict__ O) {
    __shared__ __align__(16) __bf16 kbuf[2][32][264];   // 2 x 16.5 KB
    __shared__ __align__(16) __bf16 vbuf[2][256][40];   // 2 x 20 KB
    __shared__ __align__(16) __bf16 pbuf[8][16][40];    // 10 KB, wave-private tiles

    const int tid   = threadIdx.x;
    const int w     = tid >> 5;
    const int lane  = tid & 31;
    const int qbase = blockIdx.x * 128 + w * 16;        // global token index
    const int batch = qbase >> 12;
    const int lm    = lane & 15;
    const int g16   = lane >> 4;
    const int koff  = g16 * 8;

    const __bf16* qr = Q  + (size_t)(qbase + lm) * C_DIM;
    const __bf16* Kb = K  + (size_t)batch * N_TOK * C_DIM;
    const __bf16* Vb = Vt + (size_t)batch * C_DIM * N_TOK;

    float mrow[8], lrow[8];
#pragma unroll
    for (int r = 0; r < 8; ++r) { mrow[r] = -3.0e38f; lrow[r] = 0.f; }
    v8f oacc[16];
#pragma unroll
    for (int nt = 0; nt < 16; ++nt) oacc[nt] = vzero8();

    const int NCH = N_TOK / 32;   // 128 chunks

    // Prologue: DMA chunk 0 into buffer 0 (flat->LDS addr = low 32 bits).
    if (w == 0) {
        tdm_load_2d(Kb, (unsigned)(size_t)&kbuf[0][0][0], TDM_FLAGS_K,
                    128u, (unsigned)N_TOK, 128u, 32u, 128u);
        tdm_load_2d(Vb, (unsigned)(size_t)&vbuf[0][0][0], TDM_FLAGS_V,
                    2048u, (unsigned)C_DIM, 16u, 256u, 2048u);
    }

#pragma unroll 1
    for (int ch = 0; ch < NCH; ++ch) {
        const int kt  = ch << 5;
        const int cur = ch & 1;
        if (w == 0) {
            if (ch + 1 < NCH) {
                const int nxt = cur ^ 1;
                tdm_load_2d(Kb + (size_t)(kt + 32) * C_DIM,
                            (unsigned)(size_t)&kbuf[nxt][0][0], TDM_FLAGS_K,
                            128u, (unsigned)N_TOK, 128u, 32u, 128u);
                tdm_load_2d(Vb + (kt + 32),
                            (unsigned)(size_t)&vbuf[nxt][0][0], TDM_FLAGS_V,
                            2048u, (unsigned)C_DIM, 16u, 256u, 2048u);
                __builtin_amdgcn_s_wait_tensorcnt((short)2);  // chunk ch landed
            } else {
                __builtin_amdgcn_s_wait_tensorcnt((short)0);
            }
        }
        __syncthreads();   // publish kbuf[cur] / vbuf[cur] to all waves

        // ---- S = Q K^T for two 16-key tiles (K fragments from LDS) ----
        v8f s0 = vzero8(), s1 = vzero8();
#pragma unroll
        for (int ks = 0; ks < 8; ++ks) {
            const int kb = ks * 32 + koff;
            ABfrag a, b0, b1;
            a.h[0] = *(const v8bf*)(qr + kb);
            a.h[1] = *(const v8bf*)(qr + kb + 16);
            b0.h[0] = *(const v8bf*)&kbuf[cur][lm][kb];
            b0.h[1] = *(const v8bf*)&kbuf[cur][lm][kb + 16];
            b1.h[0] = *(const v8bf*)&kbuf[cur][16 + lm][kb];
            b1.h[1] = *(const v8bf*)&kbuf[cur][16 + lm][kb + 16];
            s0 = WMMA_BF16(a.v, b0.v, s0);
            s1 = WMMA_BF16(a.v, b1.v, s1);
        }

        // ---- online softmax update (scale = C^-0.5 = 1/16) ----
        float fac[8], p0[8], p1[8];
#pragma unroll
        for (int r = 0; r < 8; ++r) {
            const float sa = s0[r] * 0.0625f;
            const float sb = s1[r] * 0.0625f;
            float mx = fmaxf(sa, sb);
            mx = fmaxf(mx, __shfl_xor(mx, 1, 32));
            mx = fmaxf(mx, __shfl_xor(mx, 2, 32));
            mx = fmaxf(mx, __shfl_xor(mx, 4, 32));
            mx = fmaxf(mx, __shfl_xor(mx, 8, 32));
            const float mnew = fmaxf(mrow[r], mx);
            fac[r] = __expf(mrow[r] - mnew);
            p0[r]  = __expf(sa - mnew);
            p1[r]  = __expf(sb - mnew);
            float rs = p0[r] + p1[r];
            rs += __shfl_xor(rs, 1, 32);
            rs += __shfl_xor(rs, 2, 32);
            rs += __shfl_xor(rs, 4, 32);
            rs += __shfl_xor(rs, 8, 32);
            lrow[r] = lrow[r] * fac[r] + rs;
            mrow[r] = mnew;
        }
#pragma unroll
        for (int nt = 0; nt < 16; ++nt)
#pragma unroll
            for (int r = 0; r < 8; ++r) oacc[nt][r] *= fac[r];

        // ---- stage P through LDS: D layout (col=lane) -> A layout (row=lane) ----
#pragma unroll
        for (int r = 0; r < 8; ++r) {
            pbuf[w][r + 8 * g16][lm]      = (__bf16)p0[r];
            pbuf[w][r + 8 * g16][16 + lm] = (__bf16)p1[r];
        }
        __syncthreads();   // orders P store->load; also K-buffer read fence
        ABfrag pa;
        pa.h[0] = *(const v8bf*)&pbuf[w][lm][koff];
        pa.h[1] = *(const v8bf*)&pbuf[w][lm][koff + 16];

        // ---- O += P * V  (V fragments from LDS, conflict-free rows) ----
#pragma unroll
        for (int nt = 0; nt < 16; ++nt) {
            const int c = nt * 16 + lm;
            ABfrag b;
            b.h[0] = *(const v8bf*)&vbuf[cur][c][koff];
            b.h[1] = *(const v8bf*)&vbuf[cur][c][koff + 16];
            oacc[nt] = WMMA_BF16(pa.v, b.v, oacc[nt]);
        }
        __syncthreads();   // all reads of buf[cur] done before its refill
    }

    // ---- finalize: O = O / l ----
    float inv[8];
#pragma unroll
    for (int r = 0; r < 8; ++r) inv[r] = 1.f / lrow[r];
#pragma unroll
    for (int nt = 0; nt < 16; ++nt) {
        const int c = nt * 16 + lm;
#pragma unroll
        for (int r = 0; r < 8; ++r) {
            const int tok = qbase + r + 8 * g16;
            O[(size_t)tok * C_DIM + c] = (__bf16)(oacc[nt][r] * inv[r]);
        }
    }
}

// ---------------------------------------------------------------------------
// Kernel 3: out projection + bias + residual, f32 output.
// ---------------------------------------------------------------------------
__global__ __launch_bounds__(256, 1)
void k_oproj(const __bf16* __restrict__ O, const __bf16* __restrict__ wto,
             const float* __restrict__ bo, const float* __restrict__ x,
             float* __restrict__ out) {
    const int tid  = threadIdx.x;
    const int w    = tid >> 5;
    const int lane = tid & 31;
    const int blockbase = blockIdx.x * 64;
    const int mt   = (w & 3) * 16;
    const int nh   = (w >> 2) * 128;
    const int lm   = lane & 15;
    const int g16  = lane >> 4;
    const int koff = g16 * 8;

    const __bf16* orow = O + (size_t)(blockbase + mt + lm) * C_DIM;

    v8f acc[8];
#pragma unroll
    for (int nt = 0; nt < 8; ++nt) acc[nt] = vzero8();

#pragma unroll 1
    for (int ks = 0; ks < 8; ++ks) {
        const int kb = ks * 32 + koff;
        ABfrag a;
        a.h[0] = *(const v8bf*)(orow + kb);
        a.h[1] = *(const v8bf*)(orow + kb + 16);
#pragma unroll
        for (int nt = 0; nt < 8; ++nt) {
            const int d = nh + nt * 16 + lm;
            ABfrag b;
            b.h[0] = *(const v8bf*)(wto + d * 256 + kb);
            b.h[1] = *(const v8bf*)(wto + d * 256 + kb + 16);
            acc[nt] = WMMA_BF16(a.v, b.v, acc[nt]);
        }
    }
#pragma unroll
    for (int nt = 0; nt < 8; ++nt) {
        const int d = nh + nt * 16 + lm;
        const float bsv = bo[d];
#pragma unroll
        for (int r = 0; r < 8; ++r) {
            const int tok = blockbase + mt + r + 8 * g16;
            const size_t off = (size_t)tok * C_DIM + d;
            out[off] = x[off] + acc[nt][r] + bsv;
        }
    }
}

// ---------------------------------------------------------------------------
extern "C" void kernel_launch(void* const* d_in, const int* in_sizes, int n_in,
                              void* d_out, int out_size, void* d_ws, size_t ws_size,
                              hipStream_t stream) {
    const float* x     = (const float*)d_in[0];
    const float* gamma = (const float*)d_in[1];
    const float* beta  = (const float*)d_in[2];
    const float* wq    = (const float*)d_in[3];
    const float* bq    = (const float*)d_in[4];
    const float* wk    = (const float*)d_in[5];
    const float* bk    = (const float*)d_in[6];
    const float* wv    = (const float*)d_in[7];
    const float* bv    = (const float*)d_in[8];
    const float* wo    = (const float*)d_in[9];
    const float* bo    = (const float*)d_in[10];

    __bf16* ws = (__bf16*)d_ws;
    __bf16* WT = ws;
    __bf16* Q  = ws + WT_ELEMS;
    __bf16* K  = Q + MAT_ELEMS;
    __bf16* Vt = K + MAT_ELEMS;
    __bf16* O  = Vt + MAT_ELEMS;

    k_prep_weights<<<1024, 256, 0, stream>>>(wq, wk, wv, wo, WT);
    k_ln_qkv<<<TOK_ALL / 64, 256, 0, stream>>>(x, gamma, beta, WT, bq, bk, bv, Q, K, Vt);
    k_attn<<<TOK_ALL / 128, 256, 0, stream>>>(Q, K, Vt, O);
    k_oproj<<<TOK_ALL / 64, 256, 0, stream>>>(O, WT + 3 * 65536, bo, x, (float*)d_out);
}
